// MambaBlock_69243462746249
// MI455X (gfx1250) — matrix operand order repaired and checked by reading, not compiled, not run on hardware
//
#include <hip/hip_runtime.h>
#include <cstdint>
#include <cstddef>

#define D_MODEL 768
#define D_INNER 1536
#define NXZ     (2*D_INNER)      // 3072
#define BATCH   2
#define SEQ     8192
#define M_ROWS  (BATCH*SEQ)      // 16384
#define ALPHA   0.9f
#define CHUNK   128
#define NCHUNK  (SEQ/CHUNK)      // 64
#define KT1     (D_MODEL/32)     // 24  k-tiles for GEMM1
#define NT1     (NXZ/16)         // 192 n-tiles for GEMM1
#define KT2     (D_INNER/32)     // 48  k-tiles for GEMM2
#define NT2     (D_MODEL/16)     // 48  n-tiles for GEMM2
#define MT      (M_ROWS/16)      // 1024 m-tiles

typedef __attribute__((ext_vector_type(16))) __bf16 bf16x16;
typedef __attribute__((ext_vector_type(8)))  float  floatx8;

__device__ __forceinline__ unsigned short f2bf(float f) {
    unsigned int u = __float_as_uint(f);
    unsigned int r = (u + 0x7FFFu + ((u >> 16) & 1u)) >> 16;  // RNE
    return (unsigned short)r;
}
__device__ __forceinline__ float bf2f(unsigned short h) {
    return __uint_as_float(((unsigned int)h) << 16);
}
__device__ __forceinline__ float silu(float v) {
    return v / (1.0f + __expf(-v));
}

// ---------------------------------------------------------------------------
// Pack activation x [M, 768] fp32 -> A-fragment order bf16.
// Record layout: ((mt*KT1 + kt)*32 + lane)*16 + j  (16 bf16 = 32B per lane)
// A 16x32 bf16 fragment (ISA 7.12.2): lane = h*16 + m;
//   pair p (j=2p,2p+1): k = (p<4 ? 2p : 16+2(p-4)) + 8h + (j&1)
// ---------------------------------------------------------------------------
__global__ void pack_a_kernel(const float* __restrict__ x,
                              unsigned short* __restrict__ ap) {
    int tid = blockIdx.x * blockDim.x + threadIdx.x;
    if (tid >= MT * KT1 * 32) return;
    int lane = tid & 31;
    int tile = tid >> 5;
    int kt = tile % KT1;
    int mt = tile / KT1;
    int mm = lane & 15, hh = lane >> 4;
    const float* src = x + (size_t)(mt * 16 + mm) * D_MODEL + kt * 32;
    unsigned short tmp[16];
#pragma unroll
    for (int j = 0; j < 16; ++j) {
        int p = j >> 1;
        int base = (p < 4) ? (2 * p) : (16 + 2 * (p - 4));
        int k = base + 8 * hh + (j & 1);
        tmp[j] = f2bf(src[k]);
    }
    uint4* dst = (uint4*)(ap + (size_t)tid * 16);
    dst[0] = ((const uint4*)tmp)[0];
    dst[1] = ((const uint4*)tmp)[1];
}

// ---------------------------------------------------------------------------
// Pack weights W [K, N] fp32 (row-major) -> B-fragment order bf16.
// Record layout: ((nt*Kt + kt)*32 + lane)*16 + j
// B 32x16 bf16 fragment: lane = h*16 + n; k = (j<8 ? j : 16+(j-8)) + 8h
// ---------------------------------------------------------------------------
__global__ void pack_w_kernel(const float* __restrict__ w,
                              unsigned short* __restrict__ wp,
                              int Kt, int N, int total) {
    int tid = blockIdx.x * blockDim.x + threadIdx.x;
    if (tid >= total) return;
    int lane = tid & 31;
    int tile = tid >> 5;
    int kt = tile % Kt;
    int nt = tile / Kt;
    int nn = lane & 15, hh = lane >> 4;
    unsigned short tmp[16];
#pragma unroll
    for (int j = 0; j < 16; ++j) {
        int v = j >> 1;
        int k = ((v < 4) ? (2 * v) : (16 + 2 * (v - 4))) + 8 * hh + (j & 1);
        tmp[j] = f2bf(w[(size_t)(kt * 32 + k) * N + nt * 16 + nn]);
    }
    uint4* dst = (uint4*)(wp + (size_t)tid * 16);
    dst[0] = ((const uint4*)tmp)[0];
    dst[1] = ((const uint4*)tmp)[1];
}

// ---------------------------------------------------------------------------
// WMMA GEMM core: 256 threads = 8 waves (2x4); wave tile 32x32 (4 accums),
// block tile 64x128. Per k-step: 2 A-frags + 2 B-frags (8x b128) -> 4 WMMAs.
// ---------------------------------------------------------------------------
#define GEMM_BODY(STORE_STMT)                                                  \
    int lane = threadIdx.x & 31;                                               \
    int wave = threadIdx.x >> 5;                                               \
    int mt0 = blockIdx.x * 4 + (wave >> 2) * 2;                                \
    int nt0 = blockIdx.y * 8 + (wave & 3) * 2;                                 \
    floatx8 c00 = {0.f,0.f,0.f,0.f,0.f,0.f,0.f,0.f};                           \
    floatx8 c01 = c00, c10 = c00, c11 = c00;                                   \
    const bf16x16* a0 = (const bf16x16*)ap + (size_t)(mt0 + 0) * Kt * 32 + lane; \
    const bf16x16* a1 = (const bf16x16*)ap + (size_t)(mt0 + 1) * Kt * 32 + lane; \
    const bf16x16* b0 = (const bf16x16*)wp + (size_t)(nt0 + 0) * Kt * 32 + lane; \
    const bf16x16* b1 = (const bf16x16*)wp + (size_t)(nt0 + 1) * Kt * 32 + lane; \
    for (int kt = 0; kt < Kt; ++kt) {                                          \
        bf16x16 fa0 = a0[kt * 32];                                             \
        bf16x16 fa1 = a1[kt * 32];                                             \
        bf16x16 fb0 = b0[kt * 32];                                             \
        bf16x16 fb1 = b1[kt * 32];                                             \
        c00 = __builtin_amdgcn_wmma_f32_16x16x32_bf16(false, fa0, false, fb0,  \
                                                      (short)0, c00, false, false); \
        c01 = __builtin_amdgcn_wmma_f32_16x16x32_bf16(false, fa0, false, fb1,  \
                                                      (short)0, c01, false, false); \
        c10 = __builtin_amdgcn_wmma_f32_16x16x32_bf16(false, fa1, false, fb0,  \
                                                      (short)0, c10, false, false); \
        c11 = __builtin_amdgcn_wmma_f32_16x16x32_bf16(false, fa1, false, fb1,  \
                                                      (short)0, c11, false, false); \
    }                                                                          \
    int hh = lane >> 4;                                                        \
    int lm = lane & 15;                                                        \
    int n0 = nt0 * 16 + lm, n1 = n0 + 16;                                      \
    float bv0 = bias[n0], bv1 = bias[n1];                                      \
    int r0 = mt0 * 16 + 8 * hh, r1 = r0 + 16;                                  \
    _Pragma("unroll")                                                          \
    for (int v = 0; v < 8; ++v) {                                              \
        STORE_STMT(r0 + v, n0, c00[v] + bv0);                                  \
        STORE_STMT(r0 + v, n1, c01[v] + bv1);                                  \
        STORE_STMT(r1 + v, n0, c10[v] + bv0);                                  \
        STORE_STMT(r1 + v, n1, c11[v] + bv1);                                  \
    }

__global__ void gemm_bf16out_kernel(const unsigned short* __restrict__ ap,
                                    const unsigned short* __restrict__ wp,
                                    const float* __restrict__ bias,
                                    unsigned short* __restrict__ out,
                                    int Kt, int N) {
#define STORE_BF(RR, NN, VAL) out[(size_t)(RR) * N + (NN)] = f2bf(VAL)
    GEMM_BODY(STORE_BF)
#undef STORE_BF
}

__global__ void gemm_f32out_kernel(const unsigned short* __restrict__ ap,
                                   const unsigned short* __restrict__ wp,
                                   const float* __restrict__ bias,
                                   float* __restrict__ out,
                                   int Kt, int N) {
#define STORE_F32(RR, NN, VAL) out[(size_t)(RR) * N + (NN)] = (VAL)
    GEMM_BODY(STORE_F32)
#undef STORE_F32
}

// ---------------------------------------------------------------------------
// Scan pass 1: per (b, channel, chunk) run conv+SiLU+EMA with h0 = 0,
// store the chunk-end state (carry). tid layout: c fastest -> coalesced xz reads.
// ---------------------------------------------------------------------------
__global__ void scan1_kernel(const unsigned short* __restrict__ xz,
                             const float* __restrict__ conv_w,
                             const float* __restrict__ conv_b,
                             float* __restrict__ carry) {
    int tid = blockIdx.x * blockDim.x + threadIdx.x;
    if (tid >= BATCH * NCHUNK * D_INNER) return;
    int c = tid % D_INNER;
    int r = tid / D_INNER;
    int chunk = r % NCHUNK;
    int b = r / NCHUNK;

    float w0 = conv_w[c * 4 + 0], w1 = conv_w[c * 4 + 1];
    float w2 = conv_w[c * 4 + 2], w3 = conv_w[c * 4 + 3];
    float cb = conv_b[c];
    int t0 = chunk * CHUNK;
    size_t rowbase = (size_t)(b * SEQ) * NXZ + c;

    float xm3 = (t0 - 3 >= 0) ? bf2f(xz[rowbase + (size_t)(t0 - 3) * NXZ]) : 0.f;
    float xm2 = (t0 - 2 >= 0) ? bf2f(xz[rowbase + (size_t)(t0 - 2) * NXZ]) : 0.f;
    float xm1 = (t0 - 1 >= 0) ? bf2f(xz[rowbase + (size_t)(t0 - 1) * NXZ]) : 0.f;
    float h = 0.f;
    for (int i = 0; i < CHUNK; ++i) {
        float x0 = bf2f(xz[rowbase + (size_t)(t0 + i) * NXZ]);
        float cv = w0 * xm3 + w1 * xm2 + w2 * xm1 + w3 * x0 + cb;
        float xc = silu(cv);
        h = ALPHA * h + (1.0f - ALPHA) * xc;
        xm3 = xm2; xm2 = xm1; xm1 = x0;
    }
    carry[tid] = h;
}

// ---------------------------------------------------------------------------
// Scan pass 2: fold chunk carries (decay 0.9^128 per chunk), replay chunk,
// fuse gating y = (xc*D + h)*silu(z) and scatter y straight into the
// A-fragment-packed bf16 buffer for GEMM2.
// ---------------------------------------------------------------------------
__global__ void scan2_kernel(const unsigned short* __restrict__ xz,
                             const float* __restrict__ conv_w,
                             const float* __restrict__ conv_b,
                             const float* __restrict__ carry,
                             const float* __restrict__ Dp,
                             unsigned short* __restrict__ ypack) {
    int tid = blockIdx.x * blockDim.x + threadIdx.x;
    if (tid >= BATCH * NCHUNK * D_INNER) return;
    int c = tid % D_INNER;
    int r = tid / D_INNER;
    int chunk = r % NCHUNK;
    int b = r / NCHUNK;

    // h0 = sum_{j<chunk} carry_j * d128^(chunk-1-j)
    const float d128 = __powf(ALPHA, (float)CHUNK);
    float h = 0.f;
    const float* cb_row = carry + (size_t)(b * NCHUNK) * D_INNER + c;
    for (int j = 0; j < chunk; ++j)
        h = h * d128 + cb_row[(size_t)j * D_INNER];

    float w0 = conv_w[c * 4 + 0], w1 = conv_w[c * 4 + 1];
    float w2 = conv_w[c * 4 + 2], w3 = conv_w[c * 4 + 3];
    float cbv = conv_b[c];
    float dpar = Dp[c];
    int t0 = chunk * CHUNK;
    size_t rowbase = (size_t)(b * SEQ) * NXZ + c;

    float xm3 = (t0 - 3 >= 0) ? bf2f(xz[rowbase + (size_t)(t0 - 3) * NXZ]) : 0.f;
    float xm2 = (t0 - 2 >= 0) ? bf2f(xz[rowbase + (size_t)(t0 - 2) * NXZ]) : 0.f;
    float xm1 = (t0 - 1 >= 0) ? bf2f(xz[rowbase + (size_t)(t0 - 1) * NXZ]) : 0.f;

    // Fragment slot for this channel (k = c within its 32-wide k-tile).
    int kt = c >> 5, kk = c & 31;
    int hh = (kk >> 3) & 1;
    int q = kk & 7;
    int p = ((kk >= 16) ? 4 : 0) + (q >> 1);
    int jslot = 2 * p + (q & 1);

    for (int i = 0; i < CHUNK; ++i) {
        int t = t0 + i;
        float x0 = bf2f(xz[rowbase + (size_t)t * NXZ]);
        float cv = w0 * xm3 + w1 * xm2 + w2 * xm1 + w3 * x0 + cbv;
        float xc = silu(cv);
        h = ALPHA * h + (1.0f - ALPHA) * xc;
        float zv = silu(bf2f(xz[rowbase + (size_t)t * NXZ + D_INNER]));
        float y = (xc * dpar + h) * zv;

        int m = b * SEQ + t;
        int mt = m >> 4, mm = m & 15;
        size_t idx = (((size_t)(mt * KT2 + kt) * 32) + (hh * 16 + mm)) * 16 + jslot;
        ypack[idx] = f2bf(y);
        xm3 = xm2; xm2 = xm1; xm1 = x0;
    }
}

// ---------------------------------------------------------------------------
extern "C" void kernel_launch(void* const* d_in, const int* in_sizes, int n_in,
                              void* d_out, int out_size, void* d_ws, size_t ws_size,
                              hipStream_t stream) {
    const float* x      = (const float*)d_in[0];
    const float* in_w   = (const float*)d_in[1];
    const float* in_b   = (const float*)d_in[2];
    const float* conv_w = (const float*)d_in[3];
    const float* conv_b = (const float*)d_in[4];
    // d_in[5..8] (xp_w, xp_b, dt_w, dt_b) are dead code in the reference.
    const float* Dp     = (const float*)d_in[9];
    const float* out_w  = (const float*)d_in[10];
    const float* out_b  = (const float*)d_in[11];
    float* out = (float*)d_out;

    char* ws = (char*)d_ws;
    size_t o = 0;
    unsigned short* xpack = (unsigned short*)(ws + o); o += (size_t)MT * KT1 * 512 * 2;   // 25.2 MB
    unsigned short* wp1   = (unsigned short*)(ws + o); o += (size_t)KT1 * NT1 * 512 * 2;  // 4.7 MB
    unsigned short* xzbuf = (unsigned short*)(ws + o); o += (size_t)M_ROWS * NXZ * 2;     // 100.7 MB
    float*          carry = (float*)(ws + o);          o += (size_t)BATCH * NCHUNK * D_INNER * 4;
    unsigned short* ypack = (unsigned short*)(ws + o); o += (size_t)MT * KT2 * 512 * 2;   // 50.3 MB
    unsigned short* wp2   = (unsigned short*)(ws + o); o += (size_t)KT2 * NT2 * 512 * 2;  // 2.4 MB
    (void)o; (void)ws_size; (void)in_sizes; (void)n_in; (void)out_size;

    // 1) pack operands to WMMA fragment order (fp32 -> bf16)
    pack_a_kernel<<<(MT * KT1 * 32) / 256, 256, 0, stream>>>(x, xpack);
    pack_w_kernel<<<(KT1 * NT1 * 32) / 256, 256, 0, stream>>>(in_w, wp1, KT1, NXZ,
                                                              KT1 * NT1 * 32);
    pack_w_kernel<<<(KT2 * NT2 * 32) / 256, 256, 0, stream>>>(out_w, wp2, KT2, D_MODEL,
                                                              KT2 * NT2 * 32);

    // 2) GEMM1: xz = x @ in_w + in_b  (bf16 out); block tile 64x128
    gemm_bf16out_kernel<<<dim3(M_ROWS / 64, NXZ / 128), 256, 0, stream>>>(
        xpack, wp1, in_b, xzbuf, KT1, NXZ);

    // 3) chunked EMA scan (conv + SiLU fused), then replay + gate into ypack
    scan1_kernel<<<(BATCH * NCHUNK * D_INNER) / 256, 256, 0, stream>>>(
        xzbuf, conv_w, conv_b, carry);
    scan2_kernel<<<(BATCH * NCHUNK * D_INNER) / 256, 256, 0, stream>>>(
        xzbuf, conv_w, conv_b, carry, Dp, ypack);

    // 4) GEMM2: out = y @ out_w + out_b (fp32 out); block tile 64x128
    gemm_f32out_kernel<<<dim3(M_ROWS / 64, D_MODEL / 128), 256, 0, stream>>>(
        ypack, wp2, out_b, out, KT2, D_MODEL);
}
